// RelativeBiasAttention_70849780515225
// MI455X (gfx1250) — matrix operand order repaired and checked by reading, not compiled, not run on hardware
//
#include <hip/hip_runtime.h>

// ---------------------------------------------------------------------------
// RelativeBiasAttention for MI455X (gfx1250, wave32, WMMA bf16 + TDM)
//   B=4, S=2048, D=1024, H=16, Dh=64
// Scratch layout in d_ws (needs ~88 MiB):
//   xb   bf16 [8192,1024]            @ 0        (16 MiB)
//   wqb/wkb/wvb/wob bf16 [1024,1024] @ 16..24M  (4 x 2 MiB)
//   Qh   bf16 [B,H,S,64]             @ 24M      (16 MiB)
//   Kh   bf16 [B,H,S,64]             @ 40M      (16 MiB)
//   Vt   bf16 [B,H,64,S]             @ 56M      (16 MiB)
//   ctx  bf16 [8192,1024]            @ 72M      (16 MiB)
// ---------------------------------------------------------------------------

#define D_MODEL 1024
#define N_HEADS 16
#define D_HEAD  64
#define BATCH   4
#define SEQ     2048
#define M_TOK   (BATCH * SEQ) // 8192

typedef __attribute__((ext_vector_type(16))) __bf16 v16bf;
typedef __attribute__((ext_vector_type(8)))  float  v8f;
typedef __attribute__((ext_vector_type(4)))  unsigned int u32x4;
typedef __attribute__((ext_vector_type(8)))  int         i32x8;
typedef __attribute__((ext_vector_type(4)))  int         i32x4;

union AB16 { uint4 q[2]; unsigned short us[16]; v16bf v; };

__device__ __forceinline__ unsigned short f2bf(float f) {
  unsigned u = __builtin_bit_cast(unsigned, f);
  u += 0x7FFFu + ((u >> 16) & 1u); // round-to-nearest-even
  return (unsigned short)(u >> 16);
}

// TDM: load one 16-row x 32-col f32 tile (row stride SEQ floats) into LDS.
// D# per CDNA5 ISA ch8: group0 = {count|flags, lds_addr, global_addr, type},
// group1 = {data_size, tensor dims, tile dims, dim0 stride}.
// This toolchain (clang-23 lane) takes the 6-arg builtin form:
// (u32x4 g0, i32x8 g1, i32x4 g2, i32x4 g3, i32x8 g4, i32 cpol).
__device__ __forceinline__ void tdm_load_bias_tile(const float* gsrc,
                                                   unsigned lds_off) {
  unsigned long long ga = (unsigned long long)(uintptr_t)gsrc;
  u32x4 g0 = { 1u,                                   // count=1 (valid, user)
               lds_off,                               // LDS byte address
               (unsigned)ga,                          // global_addr[31:0]
               (unsigned)(ga >> 32) | (2u << 30) };   // addr[56:32] | type=2
  i32x8 g1 = { (int)(2u << 16),        // data_size = 4 bytes
               (int)(2048u << 16),     // tensor_dim0 = 2048 (low 16 @ [63:48])
               (int)(16u << 16),       // tensor_dim1 = 16   (low 16 @ [95:80])
               (int)(32u << 16),       // tile_dim0 = 32     (@ [127:112])
               16,                     // tile_dim1 = 16     (@ [143:128])
               2048,                   // tensor_dim0_stride[31:0] = SEQ
               0, 0 };
  i32x4 gz4 = { 0, 0, 0, 0 };
  i32x8 gz8 = { 0, 0, 0, 0, 0, 0, 0, 0 };
  __builtin_amdgcn_tensor_load_to_lds(g0, g1, gz4, gz4, gz8, 0);
}

// --------------------------- f32 -> bf16 cast ------------------------------
__global__ void cast_f32_bf16(const float* __restrict__ in,
                              unsigned short* __restrict__ out, int n) {
  int i = blockIdx.x * blockDim.x + threadIdx.x;
  if (i < n) out[i] = f2bf(in[i]);
}

// --------------------------- WMMA GEMM -------------------------------------
// Y[m,e] = sum_k A[m,k] * W[e,k]   (A: [8192,1024] bf16, W: [1024,1024] bf16)
// mode 0: write bf16 head-major      [B,H,S,Dh]
// mode 1: write bf16 head-transposed [B,H,Dh,S]   (for V)
// mode 2: write f32  [m, e] + bias[e]             (final projection)
// One wave computes a 32(M) x 64(N) tile: 2 A-frags x 4 B-frags -> 8 WMMA
// per 32-wide K step, with explicit register double-buffering so fragment
// loads for step k+1 are in flight while step k's WMMAs issue.
__global__ __launch_bounds__(256) void gemm_bf16_wmma(
    const unsigned short* __restrict__ A, const unsigned short* __restrict__ W,
    unsigned short* __restrict__ obf, float* __restrict__ of32,
    const float* __restrict__ bias, int mode) {
  const int lane = threadIdx.x & 31;
  const int wid  = threadIdx.x >> 5;
  const int wave = blockIdx.x * (blockDim.x >> 5) + wid;
  const int TN = D_MODEL / 64; // 16 N-tiles
  const int tm = wave / TN, tn = wave % TN;
  const int m0 = tm * 32, n0 = tn * 64;
  const int row = lane & 15;
  const int hi  = (lane >> 4) & 1;

  v8f c[2][4] = {};
  const unsigned short* arow0 = A + (size_t)(m0 + row) * D_MODEL;
  const unsigned short* arow1 = arow0 + (size_t)16 * D_MODEL;

  v16bf abuf[2][2]; // [buffer][m-half]
  v16bf bbuf[2][4]; // [buffer][n-tile]

#define LOAD_A(buf, kk)                                                        \
  do {                                                                         \
    AB16 t0, t1;                                                               \
    t0.q[0] = *(const uint4*)(arow0 + (kk) + hi * 8);                          \
    t0.q[1] = *(const uint4*)(arow0 + (kk) + 16 + hi * 8);                     \
    t1.q[0] = *(const uint4*)(arow1 + (kk) + hi * 8);                          \
    t1.q[1] = *(const uint4*)(arow1 + (kk) + 16 + hi * 8);                     \
    abuf[buf][0] = t0.v; abuf[buf][1] = t1.v;                                  \
  } while (0)

#define LOAD_B(buf, kk)                                                        \
  do {                                                                         \
    _Pragma("unroll")                                                          \
    for (int t = 0; t < 4; ++t) {                                              \
      const unsigned short* wrow =                                             \
          W + (size_t)(n0 + t * 16 + row) * D_MODEL + (kk) + hi * 8;           \
      AB16 bm;                                                                 \
      bm.q[0] = *(const uint4*)(wrow);                                         \
      bm.q[1] = *(const uint4*)(wrow + 16);                                    \
      bbuf[buf][t] = bm.v;                                                     \
    }                                                                          \
  } while (0)

#define COMPUTE(buf)                                                           \
  do {                                                                         \
    _Pragma("unroll")                                                          \
    for (int t = 0; t < 4; ++t) {                                              \
      c[0][t] = __builtin_amdgcn_wmma_f32_16x16x32_bf16(                       \
          false, abuf[buf][0], false, bbuf[buf][t], (short)0, c[0][t],         \
          false, false);                                                       \
      c[1][t] = __builtin_amdgcn_wmma_f32_16x16x32_bf16(                       \
          false, abuf[buf][1], false, bbuf[buf][t], (short)0, c[1][t],         \
          false, false);                                                       \
    }                                                                          \
  } while (0)

  LOAD_A(0, 0);
  LOAD_B(0, 0);
  for (int kk = 0; kk < D_MODEL; kk += 64) {
    LOAD_A(1, kk + 32);
    LOAD_B(1, kk + 32);
    COMPUTE(0);
    if (kk + 64 < D_MODEL) {
      LOAD_A(0, kk + 64);
      LOAD_B(0, kk + 64);
    }
    COMPUTE(1);
  }
#undef LOAD_A
#undef LOAD_B
#undef COMPUTE

#pragma unroll
  for (int u = 0; u < 2; ++u) {
#pragma unroll
    for (int t = 0; t < 4; ++t) {
#pragma unroll
      for (int v = 0; v < 8; ++v) {
        int m = m0 + u * 16 + v + hi * 8; // C/D layout: VGPR v -> row v (+8 hi)
        int e = n0 + t * 16 + row;        // lane -> column
        float val = c[u][t][v];
        if (mode == 2) {
          of32[(size_t)m * D_MODEL + e] = val + bias[e];
        } else {
          int b = m / SEQ, s = m % SEQ;
          int h = e / D_HEAD, dh = e % D_HEAD;
          size_t idx = (mode == 0)
              ? ((size_t)(b * N_HEADS + h) * SEQ + s) * D_HEAD + dh
              : ((size_t)(b * N_HEADS + h) * D_HEAD + dh) * SEQ + s;
          obf[idx] = f2bf(val);
        }
      }
    }
  }
}

// --------------------------- Flash attention -------------------------------
// One wave owns 16 queries of one (b,h). Streams 32-key blocks:
//   S(16x32) = Q(16x64) @ K^T  (4 WMMA), + bias, online softmax,
//   P -> LDS (C/D layout) -> A-fragment, O(16x64) += P @ V (4 WMMA).
// The 268 MB bias stream is the HBM roofline floor: it is staged by the
// Tensor Data Mover into double-buffered LDS tiles one full key-block ahead
// (TENSORcnt pipeline depth 2), so HBM latency never sits on the softmax
// dependency chain and bias consumption becomes cheap LDS reads.
__global__ __launch_bounds__(256) void flash_bias_attn(
    const unsigned short* __restrict__ Q, const unsigned short* __restrict__ K,
    const unsigned short* __restrict__ Vt, const float* __restrict__ bias,
    unsigned short* __restrict__ ctx) {
  // per-wave: 2 x 512 f32 bias tiles + 16x33 P-transpose buffer
  __shared__ float sm[8][1024 + 544];
  const int lane = threadIdx.x & 31;
  const int wid  = threadIdx.x >> 5;
  const int wave = blockIdx.x * (blockDim.x >> 5) + wid;
  const int QT = SEQ / 16;
  const int bh = wave / QT;
  const int q0 = (wave % QT) * 16;
  const int b = bh / N_HEADS, h = bh % N_HEADS;
  const int row = lane & 15;
  const int hi  = (lane >> 4) & 1;

  float* bbf[2] = { &sm[wid][0], &sm[wid][512] };
  float* pbuf   = &sm[wid][1024];
  const unsigned bb_off[2] = { (unsigned)(uintptr_t)bbf[0],
                               (unsigned)(uintptr_t)bbf[1] };

  // Q fragments: 16x64 = two 16x32 A-frags
  v16bf qf[2];
  {
    const unsigned short* qrow = Q + ((size_t)bh * SEQ + q0 + row) * D_HEAD;
#pragma unroll
    for (int ks = 0; ks < 2; ++ks) {
      AB16 a;
      a.q[0] = *(const uint4*)(qrow + ks * 32 + hi * 16);
      a.q[1] = *(const uint4*)(qrow + ks * 32 + hi * 16 + 8);
      qf[ks] = a.v;
    }
  }

  // Kick off TDM pipeline: bias tiles for key blocks 0 and 1.
  const float* gb = bias + ((size_t)h * SEQ + q0) * SEQ; // 16-row band base
  tdm_load_bias_tile(gb + 0,  bb_off[0]);
  tdm_load_bias_tile(gb + 32, bb_off[1]);

  float mrun[8], lrun[8];
#pragma unroll
  for (int v = 0; v < 8; ++v) { mrun[v] = -3.0e38f; lrun[v] = 0.0f; }
  v8f o[4] = {};
  const float sc = 0.125f; // 1/sqrt(D_HEAD)

  for (int jb = 0; jb < SEQ / 32; ++jb) {
    const int j0 = jb * 32;
    // ---- scores: two 16x16 tiles over keys j0..j0+31
    v8f s[2] = {};
#pragma unroll
    for (int jt = 0; jt < 2; ++jt) {
      const unsigned short* krow =
          K + ((size_t)bh * SEQ + j0 + jt * 16 + row) * D_HEAD;
      if (j0 + 32 < SEQ)  // prefetch next block's K rows (128B rows)
        __builtin_prefetch(krow + 32 * D_HEAD, 0, 3);
#pragma unroll
      for (int ks = 0; ks < 2; ++ks) {
        AB16 bm; // B = K^T: lane = key n, contiguous 16 of K-dim
        bm.q[0] = *(const uint4*)(krow + ks * 32 + hi * 16);
        bm.q[1] = *(const uint4*)(krow + ks * 32 + hi * 16 + 8);
        s[jt] = __builtin_amdgcn_wmma_f32_16x16x32_bf16(
            false, qf[ks], false, bm.v, (short)0, s[jt], false, false);
      }
    }

    // ---- bias tile for this block is TDM-staged in LDS: wait for oldest
    // (in-order per wave; exactly 2 in flight -> tensorcnt<=1 retires it).
    __builtin_amdgcn_s_wait_tensorcnt(1);
    asm volatile("" ::: "memory");
    const float* bl = bbf[jb & 1];

    // ---- bias + online softmax (rows live across a 16-lane group)
#pragma unroll
    for (int v = 0; v < 8; ++v) {
      int m = v + hi * 8;
      float x0 = s[0][v] * sc + bl[m * 32 + row];
      float x1 = s[1][v] * sc + bl[m * 32 + 16 + row];
      float rm = fmaxf(x0, x1);
#pragma unroll
      for (int d = 8; d >= 1; d >>= 1) rm = fmaxf(rm, __shfl_xor(rm, d, 32));
      float mnew = fmaxf(mrun[v], rm);
      float corr = __expf(mrun[v] - mnew);
      float p0 = __expf(x0 - mnew);
      float p1 = __expf(x1 - mnew);
      float rs = p0 + p1;
#pragma unroll
      for (int d = 8; d >= 1; d >>= 1) rs += __shfl_xor(rs, d, 32);
      lrun[v] = lrun[v] * corr + rs;
      mrun[v] = mnew;
#pragma unroll
      for (int t = 0; t < 4; ++t) o[t][v] *= corr;
      pbuf[m * 33 + row] = p0;       // C/D layout -> LDS
      pbuf[m * 33 + 16 + row] = p1;
    }

    asm volatile("s_wait_dscnt 0" ::: "memory");

    // Re-issue TDM into the buffer just consumed (keeps pipeline depth 2;
    // tail iterations fetch a wrapped, never-read tile to keep the wait
    // immediate constant -- s_endpgm's implicit wait-idle drains them).
    tdm_load_bias_tile(gb + (size_t)((j0 + 64) & (SEQ - 1)), bb_off[jb & 1]);

    // ---- rebuild P(16x32) as an A-fragment (bf16) from LDS
    AB16 pa;
    {
      const float* pr = pbuf + row * 33;
#pragma unroll
      for (int j2 = 0; j2 < 8; ++j2) {
        pa.us[j2]     = f2bf(pr[hi * 16 + j2]);
        pa.us[8 + j2] = f2bf(pr[hi * 16 + 8 + j2]);
      }
    }

    // ---- O += P @ V  (V stored transposed: B-frag rows contiguous)
#pragma unroll
    for (int t = 0; t < 4; ++t) {
      const unsigned short* vrow =
          Vt + ((size_t)bh * D_HEAD + t * 16 + row) * SEQ + j0 + hi * 16;
      if (j0 + 32 < SEQ)  // prefetch next block's V span
        __builtin_prefetch(vrow + 32, 0, 3);
      AB16 bm;
      bm.q[0] = *(const uint4*)(vrow);
      bm.q[1] = *(const uint4*)(vrow + 8);
      o[t] = __builtin_amdgcn_wmma_f32_16x16x32_bf16(
          false, pa.v, false, bm.v, (short)0, o[t], false, false);
    }
  }

  // ---- epilogue: normalize by l and write ctx bf16 [B*S, D]
#pragma unroll
  for (int v = 0; v < 8; ++v) {
    float rl = 1.0f / lrun[v];
    int m = v + hi * 8;
    size_t ob = ((size_t)(b * SEQ + q0 + m)) * D_MODEL + h * D_HEAD;
#pragma unroll
    for (int t = 0; t < 4; ++t)
      ctx[ob + t * 16 + row] = f2bf(o[t][v] * rl);
  }
}

// ---------------------------------------------------------------------------
extern "C" void kernel_launch(void* const* d_in, const int* in_sizes, int n_in,
                              void* d_out, int out_size, void* d_ws,
                              size_t ws_size, hipStream_t stream) {
  const float* x    = (const float*)d_in[0];
  const float* bias = (const float*)d_in[1];
  const float* wq   = (const float*)d_in[2];
  const float* wk   = (const float*)d_in[3];
  const float* wv   = (const float*)d_in[4];
  const float* wo   = (const float*)d_in[5];
  const float* bo   = (const float*)d_in[6];
  float* out = (float*)d_out;

  char* ws = (char*)d_ws;
  unsigned short* xb  = (unsigned short*)(ws);
  unsigned short* wqb = (unsigned short*)(ws + (16ull << 20));
  unsigned short* wkb = (unsigned short*)(ws + (18ull << 20));
  unsigned short* wvb = (unsigned short*)(ws + (20ull << 20));
  unsigned short* wob = (unsigned short*)(ws + (22ull << 20));
  unsigned short* qh  = (unsigned short*)(ws + (24ull << 20));
  unsigned short* kh  = (unsigned short*)(ws + (40ull << 20));
  unsigned short* vt  = (unsigned short*)(ws + (56ull << 20));
  unsigned short* ctx = (unsigned short*)(ws + (72ull << 20));

  const int n_x = M_TOK * D_MODEL;       // 8,388,608
  const int n_w = D_MODEL * D_MODEL;     // 1,048,576

  cast_f32_bf16<<<(n_x + 255) / 256, 256, 0, stream>>>(x, xb, n_x);
  cast_f32_bf16<<<(n_w + 255) / 256, 256, 0, stream>>>(wq, wqb, n_w);
  cast_f32_bf16<<<(n_w + 255) / 256, 256, 0, stream>>>(wk, wkb, n_w);
  cast_f32_bf16<<<(n_w + 255) / 256, 256, 0, stream>>>(wv, wvb, n_w);
  cast_f32_bf16<<<(n_w + 255) / 256, 256, 0, stream>>>(wo, wob, n_w);

  // (M/32)*(D/64) = 256*16 = 4096 waves; 8 waves/block -> 512 blocks
  const int gemm_blocks = (M_TOK / 32) * (D_MODEL / 64) / 8;
  gemm_bf16_wmma<<<gemm_blocks, 256, 0, stream>>>(xb, wqb, qh, nullptr, nullptr, 0);
  gemm_bf16_wmma<<<gemm_blocks, 256, 0, stream>>>(xb, wkb, kh, nullptr, nullptr, 0);
  gemm_bf16_wmma<<<gemm_blocks, 256, 0, stream>>>(xb, wvb, vt, nullptr, nullptr, 1);

  // (B*H)*(S/16) = 64*128 = 8192 waves -> 1024 blocks
  const int fa_blocks = (BATCH * N_HEADS) * (SEQ / 16) / 8;
  flash_bias_attn<<<fa_blocks, 256, 0, stream>>>(qh, kh, vt, bias, ctx);

  gemm_bf16_wmma<<<gemm_blocks, 256, 0, stream>>>(ctx, wob, nullptr, out, bo, 2);
}
